// DeformableConvBlock_4715874091206
// MI455X (gfx1250) — matrix operand order repaired and checked by reading, not compiled, not run on hardware
//
#include <hip/hip_runtime.h>

// ---------------- types for WMMA ----------------
typedef __attribute__((ext_vector_type(16))) __bf16 v16bf;
typedef __attribute__((ext_vector_type(8)))  float  v8f;

union BFrag { v16bf v; unsigned u[8]; };

__device__ __forceinline__ unsigned short f2bf(float f) {
    unsigned u = __float_as_uint(f);
    unsigned r = u + 0x7FFFu + ((u >> 16) & 1u);   // round-to-nearest-even
    return (unsigned short)(r >> 16);
}
__device__ __forceinline__ unsigned pack2bf(float a, float b) {
    return (unsigned)f2bf(a) | ((unsigned)f2bf(b) << 16);
}

#define Hc   224
#define Wc   224
#define HWc  50176          // 224*224
#define Cin  64
#define Cout 64
#define K2   9
#define KTOT 576            // Cin*K2
#define AST  584            // padded LDS row stride (ushorts) to avoid bank conflicts

// ---------------- kernel 0: NCHW -> NHWC transpose ----------------
__global__ __launch_bounds__(256) void nchw2nhwc(const float* __restrict__ src,
                                                 float* __restrict__ dst) {
    __shared__ float tile[64 * 65];
    int blk = blockIdx.x;                  // B * (HW/64) = 2*784
    int b   = blk / 784;
    int pt  = (blk - b * 784) * 64;
    int tid = threadIdx.x;
    for (int i = tid; i < 4096; i += 256) {
        int c = i >> 6, pp = i & 63;
        tile[pp * 65 + c] = src[(size_t)(b * Cin + c) * HWc + pt + pp];
    }
    __syncthreads();
    for (int i = tid; i < 4096; i += 256) {
        int pl = i >> 6, c = i & 63;
        dst[((size_t)(b * HWc + pt + pl)) * Cin + c] = tile[pl * 65 + c];
    }
}

// ---------------- kernel 1: offset + mask conv (fp32, fused sigmoid) ----------------
__global__ __launch_bounds__(128) void offmask_conv(const float* __restrict__ xh,
                                                    const float* __restrict__ woff,
                                                    const float* __restrict__ boff,
                                                    const float* __restrict__ wmsk,
                                                    const float* __restrict__ bmsk,
                                                    float* __restrict__ off,
                                                    float* __restrict__ msk) {
    extern __shared__ char dynsm[];
    float* wl = (float*)dynsm;             // [tap][c][27] = 9*64*27 floats
    int tid = threadIdx.x;
    for (int i = tid; i < 9 * 64 * 27; i += 128) {
        int oc = i % 27;
        int tc = i / 27;
        int tap = tc >> 6, c = tc & 63;
        wl[i] = (oc < 18) ? woff[(oc * Cin + c) * 9 + tap]
                          : wmsk[((oc - 18) * Cin + c) * 9 + tap];
    }
    __syncthreads();

    int pid = blockIdx.x * 128 + tid;      // 0 .. 2*HW-1 exactly
    int b   = pid / HWc;
    int p   = pid - b * HWc;
    int hh  = p / Wc, ww = p - hh * Wc;

    float acc[27];
#pragma unroll
    for (int oc = 0; oc < 27; ++oc) acc[oc] = (oc < 18) ? boff[oc] : bmsk[oc - 18];

    for (int tap = 0; tap < 9; ++tap) {
        int iy = hh + tap / 3 - 1;
        int ix = ww + tap % 3 - 1;
        if (iy < 0 || iy >= Hc || ix < 0 || ix >= Wc) continue;
        const float* xr = xh + ((size_t)(b * Hc + iy) * Wc + ix) * Cin;
        const float* wr = wl + tap * 64 * 27;
        for (int c = 0; c < 64; c += 4) {
            float4 xv = *(const float4*)(xr + c);
#pragma unroll
            for (int j = 0; j < 4; ++j) {
                float xs = (&xv.x)[j];
                const float* w4 = wr + (c + j) * 27;
#pragma unroll
                for (int oc = 0; oc < 27; ++oc) acc[oc] = fmaf(xs, w4[oc], acc[oc]);
            }
        }
    }
    for (int kpos = 0; kpos < 9; ++kpos) {
        off[((size_t)((b * 9 + kpos) * 2 + 0)) * HWc + p] = acc[2 * kpos];
        off[((size_t)((b * 9 + kpos) * 2 + 1)) * HWc + p] = acc[2 * kpos + 1];
        msk[((size_t)(b * 9 + kpos)) * HWc + p] = 1.0f / (1.0f + __expf(-acc[18 + kpos]));
    }
}

// ---------------- kernel 2: fused bilinear sample + modulate + WMMA GEMM ----------------
__global__ __launch_bounds__(128) void dcn_main(const float* __restrict__ xh,
                                                const float* __restrict__ off,
                                                const float* __restrict__ msk,
                                                const float* __restrict__ wd,
                                                float* __restrict__ out,
                                                int ntiles) {
    extern __shared__ char dynsm[];
    unsigned short* Bs = (unsigned short*)dynsm;   // [o=64][AST]  bf16 weights
    unsigned short* As = Bs + 64 * AST;            // [m=16][AST]  bf16 sampled panel
    const int tid = threadIdx.x;

    // stage B = w_deform as bf16, layout Bs[o][kpos*64 + c]
    for (int i = tid; i < Cout * KTOT; i += 128) {
        int o = i / KTOT, r = i - o * KTOT;
        int kpos = r >> 6, c = r & 63;
        Bs[o * AST + r] = f2bf(wd[((size_t)(o * Cin + c)) * 9 + kpos]);
    }
    __syncthreads();

    const int wave = tid >> 5;
    const int lane = tid & 31;
    const int hh   = lane >> 4;     // lane half
    const int l15  = lane & 15;

    for (int t = blockIdx.x; t < ntiles; t += gridDim.x) {
        int b  = t / 3136;                 // HW/16 = 3136 tiles per batch
        int p0 = (t - b * 3136) * 16;
        int h0 = p0 / Wc;
        int w0 = p0 - h0 * Wc;             // tile = 16 consecutive x positions in one row

        // ---- phase 1: cooperative bilinear sampling into As (16 x 576 bf16) ----
        for (int task = tid; task < 576; task += 128) {
            int m    = task & 15;
            int kc   = task >> 4;          // 0..35
            int kpos = kc >> 2;            // 0..8
            int c0   = (kc & 3) << 4;      // 0,16,32,48
            int p    = p0 + m;

            float dy = off[((size_t)((b * 9 + kpos) * 2 + 0)) * HWc + p];
            float dx = off[((size_t)((b * 9 + kpos) * 2 + 1)) * HWc + p];
            float mm = msk[((size_t)(b * 9 + kpos)) * HWc + p];

            int ky = kpos / 3, kx = kpos - ky * 3;
            float py = dy + (float)(h0 - 1 + ky);
            float px = dx + (float)(w0 + m - 1 + kx);
            float y0f = floorf(py), x0f = floorf(px);
            int iy0 = (int)y0f, ix0 = (int)x0f;
            float fy = py - y0f, fx = px - x0f;

            bool vy0 = (iy0 >= 0) & (iy0 < Hc);
            bool vy1 = (iy0 + 1 >= 0) & (iy0 + 1 < Hc);
            bool vx0 = (ix0 >= 0) & (ix0 < Wc);
            bool vx1 = (ix0 + 1 >= 0) & (ix0 + 1 < Wc);

            float w00 = (vy0 & vx0) ? (1.f - fy) * (1.f - fx) * mm : 0.f;
            float w01 = (vy0 & vx1) ? (1.f - fy) * fx * mm : 0.f;
            float w10 = (vy1 & vx0) ? fy * (1.f - fx) * mm : 0.f;
            float w11 = (vy1 & vx1) ? fy * fx * mm : 0.f;

            int cy0 = min(max(iy0, 0), Hc - 1), cy1 = min(max(iy0 + 1, 0), Hc - 1);
            int cx0 = min(max(ix0, 0), Wc - 1), cx1 = min(max(ix0 + 1, 0), Wc - 1);

            const float* r00 = xh + ((size_t)(b * Hc + cy0) * Wc + cx0) * Cin + c0;
            const float* r01 = xh + ((size_t)(b * Hc + cy0) * Wc + cx1) * Cin + c0;
            const float* r10 = xh + ((size_t)(b * Hc + cy1) * Wc + cx0) * Cin + c0;
            const float* r11 = xh + ((size_t)(b * Hc + cy1) * Wc + cx1) * Cin + c0;

            unsigned short* adst = As + m * AST + kpos * 64 + c0;
#pragma unroll
            for (int cc = 0; cc < 16; cc += 4) {
                float4 a0 = *(const float4*)(r00 + cc);
                float4 a1 = *(const float4*)(r01 + cc);
                float4 a2 = *(const float4*)(r10 + cc);
                float4 a3 = *(const float4*)(r11 + cc);
                float vx = w00 * a0.x + w01 * a1.x + w10 * a2.x + w11 * a3.x;
                float vy = w00 * a0.y + w01 * a1.y + w10 * a2.y + w11 * a3.y;
                float vz = w00 * a0.z + w01 * a1.z + w10 * a2.z + w11 * a3.z;
                float vw = w00 * a0.w + w01 * a1.w + w10 * a2.w + w11 * a3.w;
                unsigned* d32 = (unsigned*)(adst + cc);
                d32[0] = pack2bf(vx, vy);
                d32[1] = pack2bf(vz, vw);
            }
        }
        __syncthreads();

        // ---- phase 2: per-wave 16x16 WMMA over K = 576 (18 steps of 32) ----
        v8f acc = {};
        const unsigned short* arow = As + l15 * AST;                 // A row M = lane&15
        const unsigned short* brow = Bs + (wave * 16 + l15) * AST;   // B col N = lane&15
        for (int kk = 0; kk < 18; ++kk) {
            int base = (kk >> 1) * 64 + ((kk & 1) << 5);             // kpos*64 + cbase
            BFrag a, bb;
#pragma unroll
            for (int d = 0; d < 4; ++d)
                a.u[d] = *(const unsigned*)(arow + base + 8 * hh + 2 * d);
#pragma unroll
            for (int d = 0; d < 4; ++d)
                a.u[4 + d] = *(const unsigned*)(arow + base + 16 + 8 * hh + 2 * d);
#pragma unroll
            for (int v = 0; v < 8; ++v)
                bb.u[v] = *(const unsigned*)(brow + base + 8 * hh + 2 * v + ((v >> 2) << 3));
            acc = __builtin_amdgcn_wmma_f32_16x16x32_bf16(
                false, a.v, false, bb.v, (short)0, acc, false, false);
        }

        // ---- store: lane owns o = wave*16 + l15, pixels p0 + 8*hh + [0..7] ----
        int o = wave * 16 + l15;
        float* op = out + ((size_t)(b * Cout + o)) * HWc + p0 + 8 * hh;
        *(float4*)(op)     = make_float4(acc[0], acc[1], acc[2], acc[3]);
        *(float4*)(op + 4) = make_float4(acc[4], acc[5], acc[6], acc[7]);
        __syncthreads();
    }
}

// ---------------- launch ----------------
extern "C" void kernel_launch(void* const* d_in, const int* in_sizes, int n_in,
                              void* d_out, int out_size, void* d_ws, size_t ws_size,
                              hipStream_t stream) {
    (void)in_sizes; (void)n_in; (void)out_size; (void)ws_size;
    const float* x        = (const float*)d_in[0];
    const float* w_offset = (const float*)d_in[1];
    const float* b_offset = (const float*)d_in[2];
    const float* w_mask   = (const float*)d_in[3];
    const float* b_mask   = (const float*)d_in[4];
    const float* w_deform = (const float*)d_in[5];
    float* out = (float*)d_out;

    // workspace layout (floats)
    float* xh  = (float*)d_ws;                    // NHWC x:   2*224*224*64 = 6,422,528
    float* off = xh + (size_t)2 * HWc * Cin;      // dy/dx:    2*9*2*HW     = 1,806,336
    float* msk = off + (size_t)2 * 9 * 2 * HWc;   // mask:     2*9*HW       =   903,168

    nchw2nhwc<<<1568, 256, 0, stream>>>(x, xh);
    offmask_conv<<<784, 128, 9 * 64 * 27 * sizeof(float), stream>>>(
        xh, w_offset, b_offset, w_mask, b_mask, off, msk);
    dcn_main<<<1568, 128, (64 + 16) * AST * sizeof(unsigned short), stream>>>(
        xh, off, msk, w_deform, out, 2 * 3136);
}
